// DecoderBlock_22643067584909
// MI455X (gfx1250) — matrix operand order repaired and checked by reading, not compiled, not run on hardware
//
#include <hip/hip_runtime.h>
#include <hip/hip_bf16.h>
#include <cstdint>

// ---------------------------------------------------------------------------
// Types
// ---------------------------------------------------------------------------
typedef __bf16 bf16;
typedef __attribute__((ext_vector_type(16))) bf16  bf16x16;
typedef __attribute__((ext_vector_type(8)))  bf16  bf16x8;
typedef __attribute__((ext_vector_type(4)))  bf16  bf16x4;
typedef __attribute__((ext_vector_type(8)))  float f32x8;
typedef __attribute__((ext_vector_type(4)))  float f32x4;
typedef __attribute__((ext_vector_type(4)))  unsigned int u32x4;
typedef __attribute__((ext_vector_type(8)))  int   i32x8;
typedef __attribute__((ext_vector_type(4)))  int   i32x4;

#define LAMBDA_INIT 0.8f
#define ATTN_SCALE  0.125f   // 1/sqrt(64)

// Tensor Data Mover availability (device pass only; host pass takes fallback)
#if defined(__HIP_DEVICE_COMPILE__) && __has_builtin(__builtin_amdgcn_tensor_load_to_lds) && __has_builtin(__builtin_amdgcn_s_wait_tensorcnt)
#define USE_TDM 1
#else
#define USE_TDM 0
#endif

__device__ __forceinline__ f32x8 wmma_bf16(bf16x16 a, bf16x16 b, f32x8 c) {
  return __builtin_amdgcn_wmma_f32_16x16x32_bf16(false, a, false, b, (short)0, c,
                                                 false, false);
}

// A-fragment (16x32): lane holds row = lane%16; e<8 -> K=k0+8*half+e,
// e>=8 -> K=k0+16+8*half+(e-8). Works for LDS or global bf16 rows.
__device__ __forceinline__ bf16x16 afrag(const bf16* row, int k0, int half) {
  bf16x8 lo = *(const bf16x8*)(row + k0 + 8 * half);
  bf16x8 hi = *(const bf16x8*)(row + k0 + 16 + 8 * half);
  bf16x16 r;
#pragma unroll
  for (int i = 0; i < 8; ++i) { r[i] = lo[i]; r[i + 8] = hi[i]; }
  return r;
}

// B-fragment (32x16), stored [n][k]: lane holds col n = lane%16;
// e<8 -> K=k0+8*half+e, e>=8 -> K=k0+16+8*half+(e-8)
__device__ __forceinline__ bf16x16 bfrag(const bf16* row, int k0, int half) {
  bf16x8 lo = *(const bf16x8*)(row + k0 + 8 * half);
  bf16x8 hi = *(const bf16x8*)(row + k0 + 16 + 8 * half);
  bf16x16 r;
#pragma unroll
  for (int i = 0; i < 8; ++i) { r[i] = lo[i]; r[i + 8] = hi[i]; }
  return r;
}

#if USE_TDM
// 2-D bf16 tile -> LDS via Tensor Data Mover. One instruction per call,
// issued per-wave; completion tracked with TENSORcnt.
__device__ __forceinline__ void tdm_load_2d_bf16(unsigned lds_off, const bf16* g,
                                                 unsigned rowElems, unsigned rows,
                                                 unsigned strideElems) {
  unsigned long long ga = (unsigned long long)(uintptr_t)g;
  u32x4 g0;
  g0[0] = 1u;                                            // count=1, user D#
  g0[1] = lds_off;                                       // lds_addr (bytes)
  g0[2] = (unsigned)ga;                                  // global_addr[31:0]
  g0[3] = (unsigned)((ga >> 32) & 0x01FFFFFFu) | (2u << 30); // [56:32] | type=2
  i32x8 g1;
  g1[0] = 1 << 16;                                       // data_size=2B, mask=0
  g1[1] = (int)((rowElems & 0xFFFFu) << 16);             // tensor_dim0[15:0]
  g1[2] = (int)(((rowElems >> 16) & 0xFFFFu) | ((rows & 0xFFFFu) << 16));
  g1[3] = (int)(((rows >> 16) & 0xFFFFu) | ((rowElems & 0xFFFFu) << 16)); // tile_dim0
  g1[4] = (int)(rows & 0xFFFFu);                         // tile_dim1 (tile_dim2=0)
  g1[5] = (int)strideElems;                              // tensor_dim0_stride[31:0]
  g1[6] = 0;
  g1[7] = 0;
  i32x4 z4 = {0, 0, 0, 0};
#if __clang_major__ >= 23
  i32x8 z8 = {0, 0, 0, 0, 0, 0, 0, 0};
  __builtin_amdgcn_tensor_load_to_lds(g0, g1, z4, z4, z8, 0);
#else
  __builtin_amdgcn_tensor_load_to_lds(g0, g1, z4, z4, 0);
#endif
}
#endif

// Manual fallback: copy 128 rows x 32 bf16 (64B/row) into dst[128][32]
__device__ __forceinline__ void stage_panel(bf16 (*dst)[32], const bf16* src, int K) {
  int t = threadIdx.x;
  int row = t >> 1, c = (t & 1) << 4;
  const bf16* p = src + (size_t)row * K + c;
  bf16x8 v0 = *(const bf16x8*)p;
  bf16x8 v1 = *(const bf16x8*)(p + 8);
  *(bf16x8*)&dst[row][c] = v0;
  *(bf16x8*)&dst[row][c + 8] = v1;
}

// ---------------------------------------------------------------------------
// GEMM: C[M,N] = A[M,K] @ Wt[N,K]^T + bias[N]
// A, Wt bf16 (both K-major); C fp32 or bf16. 256 threads, tile 128x128x32,
// double-buffered LDS staged by the TDM (wave 0) when available.
// ---------------------------------------------------------------------------
__global__ __launch_bounds__(256) void gemm_bf16_kernel(
    const bf16* __restrict__ A, const bf16* __restrict__ Wt,
    const float* __restrict__ bias, void* __restrict__ Cout,
    int M, int N, int K, int out_bf16) {
  __shared__ alignas(16) bf16 As[2][128][32];
  __shared__ alignas(16) bf16 Bs[2][128][32];

  const int tid  = threadIdx.x;
  const int lane = tid & 31, w = tid >> 5;
  const int wm = w >> 2, wn = w & 3;
  const int half = lane >> 4, l16 = lane & 15;
  const int rowBase = blockIdx.y * 128;
  const int colBase = blockIdx.x * 128;
  const bf16* Abase = A + (size_t)rowBase * K;
  const bf16* Bbase = Wt + (size_t)colBase * K;

  f32x8 acc[4][2] = {};
  const int ksteps = K >> 5;

  // prologue: stage tile 0 into buffer 0
#if USE_TDM
  if (w == 0) {
    tdm_load_2d_bf16((unsigned)(uintptr_t)&As[0][0][0], Abase, 32, 128, (unsigned)K);
    tdm_load_2d_bf16((unsigned)(uintptr_t)&Bs[0][0][0], Bbase, 32, 128, (unsigned)K);
  }
#else
  stage_panel(As[0], Abase, K);
  stage_panel(Bs[0], Bbase, K);
#endif

  for (int s = 0; s < ksteps; ++s) {
    const int cur = s & 1;
    const bool hasNext = (s + 1) < ksteps;
    if (hasNext) {
      const int k1 = (s + 1) << 5;
#if USE_TDM
      if (w == 0) {
        tdm_load_2d_bf16((unsigned)(uintptr_t)&As[cur ^ 1][0][0], Abase + k1, 32, 128, (unsigned)K);
        tdm_load_2d_bf16((unsigned)(uintptr_t)&Bs[cur ^ 1][0][0], Bbase + k1, 32, 128, (unsigned)K);
      }
#else
      stage_panel(As[cur ^ 1], Abase + k1, K);
      stage_panel(Bs[cur ^ 1], Bbase + k1, K);
      __builtin_prefetch(Abase + (size_t)(tid >> 1) * K + k1 + 32, 0, 1);
#endif
    }
#if USE_TDM
    // retire the current tile's pair (the next pair may remain in flight)
    if (hasNext) __builtin_amdgcn_s_wait_tensorcnt(2);
    else         __builtin_amdgcn_s_wait_tensorcnt(0);
#endif
    __syncthreads();   // current buffer visible to every wave

    bf16x16 af[4], bfv[2];
#pragma unroll
    for (int mt = 0; mt < 4; ++mt)
      af[mt] = afrag(&As[cur][wm * 64 + mt * 16 + l16][0], 0, half);
#pragma unroll
    for (int nt = 0; nt < 2; ++nt)
      bfv[nt] = bfrag(&Bs[cur][wn * 32 + nt * 16 + l16][0], 0, half);
#pragma unroll
    for (int mt = 0; mt < 4; ++mt)
#pragma unroll
      for (int nt = 0; nt < 2; ++nt)
        acc[mt][nt] = wmma_bf16(af[mt], bfv[nt], acc[mt][nt]);
    __syncthreads();   // everyone done reading before buffer is re-staged
  }

  // epilogue: C-layout row = e + 8*half, col = l16
#pragma unroll
  for (int mt = 0; mt < 4; ++mt)
#pragma unroll
    for (int nt = 0; nt < 2; ++nt)
#pragma unroll
      for (int e = 0; e < 8; ++e) {
        int row = rowBase + wm * 64 + mt * 16 + e + 8 * half;
        int col = colBase + wn * 32 + nt * 16 + l16;
        float v = acc[mt][nt][e] + bias[col];
        if (out_bf16) ((bf16*)Cout)[(size_t)row * N + col] = (bf16)v;
        else          ((float*)Cout)[(size_t)row * N + col] = v;
      }
}

// ---------------------------------------------------------------------------
// Differential attention, bf16 in / bf16 out, flash-style.
// grid = (SQ/64, H, B), block = 256 (waves 0-3: group0, 4-7: group1)
// Q/K: [B*S][2048] col = g*1024 + head*64 + d ;  V/O: [B*S][2048] col = head*128 + d2
// ---------------------------------------------------------------------------
struct AttnSmemLoop { bf16 Kt[2][64][64]; bf16 VtT[128][64]; }; // 32 KB
struct AttnSmemEpi  { float O1s[4][16][128]; };                 // 32 KB
union  alignas(16) AttnSmem { AttnSmemLoop a; AttnSmemEpi b; };

__global__ __launch_bounds__(256) void diff_attn_kernel(
    const bf16* __restrict__ Qp, const bf16* __restrict__ Kp,
    const bf16* __restrict__ Vp,
    const float* __restrict__ lq1, const float* __restrict__ lk1,
    const float* __restrict__ lq2, const float* __restrict__ lk2,
    const float* __restrict__ gvec, bf16* __restrict__ O,
    int SKlen, int causal) {
  __shared__ AttnSmem sm;
  __shared__ alignas(16) bf16 Pw[8][16][64];
  __shared__ float s_lam;

  const int b = blockIdx.z, head = blockIdx.y;
  const int q0 = blockIdx.x * 64;
  const int tid = threadIdx.x, lane = tid & 31, w = tid >> 5;
  const int grp = w >> 2, qs = w & 3;
  const int half = lane >> 4, l16 = lane & 15;

  if (tid == 0) {
    float d1 = 0.f, d2 = 0.f;
    for (int i = 0; i < 64; ++i) { d1 += lq1[i] * lk1[i]; d2 += lq2[i] * lk2[i]; }
    s_lam = __expf(d1) - __expf(d2) + LAMBDA_INIT;
  }
  __syncthreads();
  const float lam = s_lam;

  const int qrow = q0 + qs * 16 + l16;
  const bf16* qptr = Qp + ((size_t)(b * 1024 + qrow)) * 2048 + grp * 1024 + head * 64;
  const bf16x16 qf0 = afrag(qptr, 0, half);
  const bf16x16 qf1 = afrag(qptr, 32, half);

  f32x8 acc[8] = {};
  float mrow[8], lrow[8];
#pragma unroll
  for (int e = 0; e < 8; ++e) { mrow[e] = -3.0e38f; lrow[e] = 0.f; }

  const int nsteps = causal ? (q0 / 64 + 1) : (SKlen / 64);
  for (int step = 0; step < nsteps; ++step) {
    const int k0g = step * 64;
    // ---- stage K tiles for both groups (64x64 bf16 each) ----
#if USE_TDM
    if (w == 0) {
      tdm_load_2d_bf16((unsigned)(uintptr_t)&sm.a.Kt[0][0][0],
                       Kp + ((size_t)(b * 1024 + k0g)) * 2048 + head * 64, 64, 64, 2048);
      tdm_load_2d_bf16((unsigned)(uintptr_t)&sm.a.Kt[1][0][0],
                       Kp + ((size_t)(b * 1024 + k0g)) * 2048 + 1024 + head * 64, 64, 64, 2048);
    }
#else
#pragma unroll
    for (int i = 0; i < 2; ++i) {
      int e = (tid + i * 256) * 16;              // 8192 bf16 elements
      int g2 = e >> 12, r = (e >> 6) & 63, c = e & 63;
      const bf16* p = Kp + ((size_t)(b * 1024 + k0g + r)) * 2048 + g2 * 1024 + head * 64 + c;
      bf16x8 v0 = *(const bf16x8*)p;
      bf16x8 v1 = *(const bf16x8*)(p + 8);
      *(bf16x8*)&sm.a.Kt[g2][r][c] = v0;
      *(bf16x8*)&sm.a.Kt[g2][r][c + 8] = v1;
    }
#endif
    // ---- stage V transposed: VtT[d2][t] (64x128 bf16) ----
#pragma unroll
    for (int i = 0; i < 4; ++i) {
      int e = (tid + i * 256) * 8;               // 8192 bf16 elements
      int r = e >> 7, c = e & 127;
      bf16x8 v = *(const bf16x8*)(Vp + ((size_t)(b * 1024 + k0g + r)) * 2048 + head * 128 + c);
#pragma unroll
      for (int j = 0; j < 8; ++j) sm.a.VtT[c + j][r] = v[j];
    }
#if USE_TDM
    __builtin_amdgcn_s_wait_tensorcnt(0);
#endif
    __syncthreads();

    // scores S[16x64] = Q(16x64) @ K^T (2 chained WMMAs per 16-col tile)
    f32x8 s[4];
#pragma unroll
    for (int nt = 0; nt < 4; ++nt) {
      const bf16* krow = &sm.a.Kt[grp][nt * 16 + l16][0];
      f32x8 z = {};
      z = wmma_bf16(qf0, bfrag(krow, 0, half), z);
      z = wmma_bf16(qf1, bfrag(krow, 32, half), z);
      s[nt] = z;
    }

    // online softmax
#pragma unroll
    for (int e = 0; e < 8; ++e) {
      const int rloc = e + 8 * half;
      const int rowg = q0 + qs * 16 + rloc;
      float sv[4], mx = -3.0e38f;
#pragma unroll
      for (int nt = 0; nt < 4; ++nt) {
        float t2 = s[nt][e] * ATTN_SCALE;
        int col = k0g + nt * 16 + l16;
        if (causal && col > rowg) t2 = -1.0e30f;
        sv[nt] = t2;
        mx = fmaxf(mx, t2);
      }
#pragma unroll
      for (int off = 1; off < 16; off <<= 1) mx = fmaxf(mx, __shfl_xor(mx, off, 32));
      float mnew = fmaxf(mrow[e], mx);
      float alpha = __expf(mrow[e] - mnew);
      float psum = 0.f;
#pragma unroll
      for (int nt = 0; nt < 4; ++nt) {
        float pp = __expf(sv[nt] - mnew);
        psum += pp;
        Pw[w][rloc][nt * 16 + l16] = (bf16)pp;
      }
#pragma unroll
      for (int off = 1; off < 16; off <<= 1) psum += __shfl_xor(psum, off, 32);
      lrow[e] = lrow[e] * alpha + psum;
      mrow[e] = mnew;
#pragma unroll
      for (int vt = 0; vt < 8; ++vt) acc[vt][e] *= alpha;
    }
    __syncthreads();

    // O += P @ V
    const bf16* prow = &Pw[w][l16][0];
    const bf16x16 pa0 = afrag(prow, 0, half);
    const bf16x16 pa1 = afrag(prow, 32, half);
#pragma unroll
    for (int vt = 0; vt < 8; ++vt) {
      const bf16* vrow = &sm.a.VtT[vt * 16 + l16][0];
      acc[vt] = wmma_bf16(pa0, bfrag(vrow, 0, half), acc[vt]);
      acc[vt] = wmma_bf16(pa1, bfrag(vrow, 32, half), acc[vt]);
    }
    __syncthreads();
  }

  // combine groups: O = O0/l0 - lam*O1/l1 , per-head RMSNorm, *g, *(1-li)
  if (grp == 1) {
#pragma unroll
    for (int e = 0; e < 8; ++e) {
      float inv = lam / lrow[e];
      int rloc = e + 8 * half;
#pragma unroll
      for (int vt = 0; vt < 8; ++vt)
        sm.b.O1s[qs][rloc][vt * 16 + l16] = acc[vt][e] * inv;
    }
  }
  __syncthreads();
  if (grp == 0) {
#pragma unroll
    for (int e = 0; e < 8; ++e) {
      float inv = 1.f / lrow[e];
      int rloc = e + 8 * half;
      float ss = 0.f, ov[8];
#pragma unroll
      for (int vt = 0; vt < 8; ++vt) {
        float o = acc[vt][e] * inv - sm.b.O1s[qs][rloc][vt * 16 + l16];
        ov[vt] = o;
        ss += o * o;
      }
#pragma unroll
      for (int off = 1; off < 16; off <<= 1) ss += __shfl_xor(ss, off, 32);
      float rn = rsqrtf(ss * (1.f / 128.f) + 1e-6f);
      int rowg2 = q0 + qs * 16 + rloc;
      bf16* orow = O + ((size_t)(b * 1024 + rowg2)) * 2048 + head * 128;
#pragma unroll
      for (int vt = 0; vt < 8; ++vt) {
        int col = vt * 16 + l16;
        orow[col] = (bf16)(ov[vt] * rn * gvec[col] * (1.0f - LAMBDA_INIT));
      }
    }
  }
}

// ---------------------------------------------------------------------------
// Weight prep: fp32 [K][N] -> bf16 [N][K] (LDS-tiled transpose, coalesced)
// ---------------------------------------------------------------------------
__global__ __launch_bounds__(256) void transpose_cvt_kernel(
    const float* __restrict__ W, bf16* __restrict__ Wt, int K, int N) {
  __shared__ float tile[32][33];
  const int n0 = blockIdx.x * 32, k0 = blockIdx.y * 32;
  const int tx = threadIdx.x, ty = threadIdx.y;   // 32 x 8
#pragma unroll
  for (int j = 0; j < 4; ++j)
    tile[ty * 4 + j][tx] = W[(size_t)(k0 + ty * 4 + j) * N + n0 + tx];
  __syncthreads();
#pragma unroll
  for (int j = 0; j < 4; ++j)
    Wt[(size_t)(n0 + ty * 4 + j) * K + k0 + tx] = (bf16)tile[tx][ty * 4 + j];
}

// ---------------------------------------------------------------------------
// Elementwise helpers
// ---------------------------------------------------------------------------
__global__ __launch_bounds__(256) void ln_kernel(
    const float* __restrict__ x, const float* __restrict__ g,
    const float* __restrict__ bsh, bf16* __restrict__ out) {
  const int t = blockIdx.x;
  const float* xr = x + (size_t)t * 1024;
  bf16* orow = out + (size_t)t * 1024;
  const int tid = threadIdx.x;
  f32x4 v = *(const f32x4*)(xr + tid * 4);
  float s = v[0] + v[1] + v[2] + v[3];
  float sq = v[0] * v[0] + v[1] * v[1] + v[2] * v[2] + v[3] * v[3];
#pragma unroll
  for (int off = 1; off < 32; off <<= 1) {
    s += __shfl_xor(s, off, 32);
    sq += __shfl_xor(sq, off, 32);
  }
  __shared__ float ssum[8], ssq[8], sm_, sv_;
  if ((tid & 31) == 0) { ssum[tid >> 5] = s; ssq[tid >> 5] = sq; }
  __syncthreads();
  if (tid == 0) {
    float a = 0.f, b2 = 0.f;
    for (int i = 0; i < 8; ++i) { a += ssum[i]; b2 += ssq[i]; }
    float m = a * (1.f / 1024.f);
    sm_ = m;
    sv_ = b2 * (1.f / 1024.f) - m * m;
  }
  __syncthreads();
  float m = sm_, inv = rsqrtf(sv_ + 1e-5f);
  bf16x4 o;
#pragma unroll
  for (int j = 0; j < 4; ++j)
    o[j] = (bf16)((v[j] - m) * inv * g[tid * 4 + j] + bsh[tid * 4 + j]);
  *(bf16x4*)(orow + tid * 4) = o;
}

__global__ void cvt_bf16_kernel(bf16* __restrict__ d, const float* __restrict__ s, int n) {
  int i = blockIdx.x * blockDim.x + threadIdx.x;
  if (i < n) d[i] = (bf16)s[i];
}
__global__ void copy_kernel(float* __restrict__ d, const float* __restrict__ s, int n) {
  int i = blockIdx.x * blockDim.x + threadIdx.x;
  if (i < n) d[i] = s[i];
}
__global__ void add_inplace_kernel(float* __restrict__ d, const float* __restrict__ s, int n) {
  int i = blockIdx.x * blockDim.x + threadIdx.x;
  if (i < n) d[i] += s[i];
}
__global__ void add_out_kernel(float* __restrict__ o, const float* __restrict__ a,
                               const float* __restrict__ b2, int n) {
  int i = blockIdx.x * blockDim.x + threadIdx.x;
  if (i < n) o[i] = a[i] + b2[i];
}
__global__ void silu_mul_kernel(const bf16* __restrict__ gt, const bf16* __restrict__ vt,
                                bf16* __restrict__ o, int n) {
  int i = blockIdx.x * blockDim.x + threadIdx.x;
  if (i < n) {
    float gv = (float)gt[i];
    o[i] = (bf16)(gv / (1.f + __expf(-gv)) * (float)vt[i]);
  }
}

// ---------------------------------------------------------------------------
// Orchestration
// ---------------------------------------------------------------------------
extern "C" void kernel_launch(void* const* d_in, const int* in_sizes, int n_in,
                              void* d_out, int out_size, void* d_ws, size_t ws_size,
                              hipStream_t stream) {
  (void)in_sizes; (void)n_in; (void)out_size; (void)ws_size;
  const float* X    = (const float*)d_in[0];
  const float* ENC  = (const float*)d_in[1];
  // d_in[2] = target_mask (causal tril) — computed analytically in-kernel
  const float* LN1G = (const float*)d_in[3],  *LN1B = (const float*)d_in[4];
  const float* LN2G = (const float*)d_in[5],  *LN2B = (const float*)d_in[6];
  const float* LN3G = (const float*)d_in[7],  *LN3B = (const float*)d_in[8];
  const float* SAWQ = (const float*)d_in[9],  *SABQ = (const float*)d_in[10];
  const float* SAWK = (const float*)d_in[11], *SABK = (const float*)d_in[12];
  const float* SAWV = (const float*)d_in[13], *SABV = (const float*)d_in[14];
  const float* SAWO = (const float*)d_in[15], *SABO = (const float*)d_in[16];
  const float* SAL1Q = (const float*)d_in[17], *SAL1K = (const float*)d_in[18];
  const float* SAL2Q = (const float*)d_in[19], *SAL2K = (const float*)d_in[20];
  const float* SAG  = (const float*)d_in[21];
  const float* CAWQ = (const float*)d_in[22], *CABQ = (const float*)d_in[23];
  const float* CAWK = (const float*)d_in[24], *CABK = (const float*)d_in[25];
  const float* CAWV = (const float*)d_in[26], *CABV = (const float*)d_in[27];
  const float* CAWO = (const float*)d_in[28], *CABO = (const float*)d_in[29];
  const float* CAL1Q = (const float*)d_in[30], *CAL1K = (const float*)d_in[31];
  const float* CAL2Q = (const float*)d_in[32], *CAL2K = (const float*)d_in[33];
  const float* CAG  = (const float*)d_in[34];
  const float* FW1 = (const float*)d_in[35], *FB1 = (const float*)d_in[36];
  const float* FWG = (const float*)d_in[37], *FBG = (const float*)d_in[38];
  const float* FWV = (const float*)d_in[39], *FBV = (const float*)d_in[40];
  const float* FW2 = (const float*)d_in[41], *FB2 = (const float*)d_in[42];
  float* OUT = (float*)d_out;

  const size_t T = 2048;   // B*SQ tokens
  char* p = (char*)d_ws;
  auto carve = [&](size_t bytes) { char* r = p; p += (bytes + 255) & ~(size_t)255; return r; };

  float* bufX = (float*)carve(T * 1024 * 4);
  float* bufY = (float*)carve(T * 1024 * 4);
  bf16* bufH = (bf16*)carve(T * 1024 * 2);
  bf16* encB = (bf16*)carve(T * 1024 * 2);
  bf16* bufQ = (bf16*)carve(T * 2048 * 2);   // } contiguous: also FFN T1
  bf16* bufK = (bf16*)carve(T * 2048 * 2);   // }
  bf16* bufV = (bf16*)carve(T * 2048 * 2);   // } contiguous: also FFN T2
  bf16* bufO = (bf16*)carve(T * 2048 * 2);   // }
  bf16* bufA = (bf16*)carve(T * 4096 * 2);
  bf16* T1 = bufQ;
  bf16* T2 = bufV;

  bf16* saWqT = (bf16*)carve(2048 * 1024 * 2);
  bf16* saWkT = (bf16*)carve(2048 * 1024 * 2);
  bf16* saWvT = (bf16*)carve(2048 * 1024 * 2);
  bf16* saWoT = (bf16*)carve(1024 * 2048 * 2);
  bf16* caWqT = (bf16*)carve(2048 * 1024 * 2);
  bf16* caWkT = (bf16*)carve(2048 * 1024 * 2);
  bf16* caWvT = (bf16*)carve(2048 * 1024 * 2);
  bf16* caWoT = (bf16*)carve(1024 * 2048 * 2);
  bf16* fW1T  = (bf16*)carve(4096 * 1024 * 2);
  bf16* fWgT  = (bf16*)carve((size_t)4096 * 4096 * 2);
  bf16* fWvT  = (bf16*)carve((size_t)4096 * 4096 * 2);
  bf16* fW2T  = (bf16*)carve(1024 * 4096 * 2);

  auto tw = [&](const float* W, bf16* Wt, int K, int N) {
    transpose_cvt_kernel<<<dim3(N / 32, K / 32), dim3(32, 8), 0, stream>>>(W, Wt, K, N);
  };
  auto gemm = [&](const bf16* A, const bf16* Wt, const float* bias, void* C,
                  int M, int N, int K, int outBf) {
    gemm_bf16_kernel<<<dim3(N / 128, M / 128), 256, 0, stream>>>(A, Wt, bias, C, M, N, K, outBf);
  };
  auto ew = [&](int n) { return dim3((n + 255) / 256); };
  const int nTok = (int)(T * 1024);
  const dim3 ag(16, 16, 2);   // (SQ/64, H, B)

  // ---- one-time per launch: weights -> transposed bf16, enc -> bf16 ----
  tw(SAWQ, saWqT, 1024, 2048); tw(SAWK, saWkT, 1024, 2048);
  tw(SAWV, saWvT, 1024, 2048); tw(SAWO, saWoT, 2048, 1024);
  tw(CAWQ, caWqT, 1024, 2048); tw(CAWK, caWkT, 1024, 2048);
  tw(CAWV, caWvT, 1024, 2048); tw(CAWO, caWoT, 2048, 1024);
  tw(FW1, fW1T, 1024, 4096);   tw(FWG, fWgT, 4096, 4096);
  tw(FWV, fWvT, 4096, 4096);   tw(FW2, fW2T, 4096, 1024);
  cvt_bf16_kernel<<<ew(nTok), 256, 0, stream>>>(encB, ENC, nTok);
  copy_kernel<<<ew(nTok), 256, 0, stream>>>(bufX, X, nTok);

  // ---- self differential attention ----
  ln_kernel<<<dim3((int)T), 256, 0, stream>>>(bufX, LN1G, LN1B, bufH);
  gemm(bufH, saWqT, SABQ, bufQ, (int)T, 2048, 1024, 1);
  gemm(bufH, saWkT, SABK, bufK, (int)T, 2048, 1024, 1);
  gemm(bufH, saWvT, SABV, bufV, (int)T, 2048, 1024, 1);
  diff_attn_kernel<<<ag, 256, 0, stream>>>(bufQ, bufK, bufV, SAL1Q, SAL1K,
                                           SAL2Q, SAL2K, SAG, bufO, 1024, 1);
  gemm(bufO, saWoT, SABO, bufY, (int)T, 1024, 2048, 0);
  add_inplace_kernel<<<ew(nTok), 256, 0, stream>>>(bufX, bufY, nTok);

  // ---- cross differential attention (K/V from raw encoder_out) ----
  ln_kernel<<<dim3((int)T), 256, 0, stream>>>(bufX, LN2G, LN2B, bufH);
  gemm(bufH, caWqT, CABQ, bufQ, (int)T, 2048, 1024, 1);
  gemm(encB, caWkT, CABK, bufK, (int)T, 2048, 1024, 1);
  gemm(encB, caWvT, CABV, bufV, (int)T, 2048, 1024, 1);
  diff_attn_kernel<<<ag, 256, 0, stream>>>(bufQ, bufK, bufV, CAL1Q, CAL1K,
                                           CAL2Q, CAL2K, CAG, bufO, 1024, 0);
  gemm(bufO, caWoT, CABO, bufY, (int)T, 1024, 2048, 0);
  add_inplace_kernel<<<ew(nTok), 256, 0, stream>>>(bufX, bufY, nTok);

  // ---- SwiGLU-style FFN ----
  ln_kernel<<<dim3((int)T), 256, 0, stream>>>(bufX, LN3G, LN3B, bufH);
  gemm(bufH, fW1T, FB1, bufA, (int)T, 4096, 1024, 1);   // a
  gemm(bufA, fWgT, FBG, T1, (int)T, 4096, 4096, 1);     // a@Wg
  gemm(bufA, fWvT, FBV, T2, (int)T, 4096, 4096, 1);     // a@Wv
  const int nFF = (int)(T * 4096);
  silu_mul_kernel<<<ew(nFF), 256, 0, stream>>>(T1, T2, bufA, nFF);
  gemm(bufA, fW2T, FB2, bufY, (int)T, 1024, 4096, 0);
  add_out_kernel<<<ew(nTok), 256, 0, stream>>>(OUT, bufX, bufY, nTok);
}
